// EdgeAwareDirectGNNPredictor_88940182765954
// MI455X (gfx1250) — matrix operand order repaired and checked by reading, not compiled, not run on hardware
//
#include <hip/hip_runtime.h>
#include <hip/hip_bf16.h>

#define HID 64
#define NLAYERS 6
#define N_NODES_C 100000
#define TILES 4   // 16-row tiles per wave per kernel launch

typedef __bf16 bf16_t;
typedef __attribute__((ext_vector_type(16))) __bf16 bf16x16;
typedef __attribute__((ext_vector_type(8)))  __bf16 bf16x8;
typedef __attribute__((ext_vector_type(8)))  float  f32x8;

// ---------------------------------------------------------------------------
// WMMA helper: D = A(16x32 bf16) * B(32x16 bf16) + C(16x16 f32)
// ---------------------------------------------------------------------------
__device__ __forceinline__ f32x8 wmma_bf16(bf16x16 a, bf16x16 b, f32x8 c) {
  return __builtin_amdgcn_wmma_f32_16x16x32_bf16(false, a, false, b, (short)0, c,
                                                 false, false);
}

__device__ __forceinline__ bf16x16 cat8(bf16x8 lo, bf16x8 hi) {
  bf16x16 r;
#pragma unroll
  for (int i = 0; i < 8; ++i) { r[i] = lo[i]; r[i + 8] = hi[i]; }
  return r;
}

// A-matrix fragment for 16-bit 16x32 tile (ISA 7.12.2):
//   row M = lane&15, half = lane>>4
//   elems 0..7  -> K = koff + 8*half + [0..8)
//   elems 8..15 -> K = koff + 16 + 8*half + [0..8)
__device__ __forceinline__ bf16x16 load_a_row(const bf16_t* rowp, int koff, int half) {
  bf16x8 lo = *(const bf16x8*)(rowp + koff + 8 * half);
  bf16x8 hi = *(const bf16x8*)(rowp + koff + 16 + 8 * half);
  return cat8(lo, hi);
}

// B-matrix fragment from pre-packed layout: tile = 32 lanes x 16 bf16 (32B/lane)
__device__ __forceinline__ bf16x16 load_b_packed(const bf16_t* tile_base, int lane) {
  const bf16_t* p = tile_base + lane * 16;
  bf16x8 lo = *(const bf16x8*)p;
  bf16x8 hi = *(const bf16x8*)(p + 8);
  return cat8(lo, hi);
}

// gfx1250 async global->LDS copy (16B per lane), tracked by ASYNCcnt.
__device__ __forceinline__ void async_copy16(void* lds_dst, const void* gsrc) {
  unsigned lds_off = (unsigned)(size_t)lds_dst;  // low 32 bits = LDS offset
  asm volatile("global_load_async_to_lds_b128 %0, %1, off"
               :: "v"(lds_off), "v"(gsrc) : "memory");
}
__device__ __forceinline__ void wait_async0() {
  asm volatile("s_wait_asynccnt 0x0" ::: "memory");
}

__device__ __forceinline__ float silu_f(float x) {
  return x * (1.0f / (1.0f + __expf(-x)));
}

__device__ __forceinline__ unsigned encf(float f) {
  unsigned u = __float_as_uint(f);
  return (u & 0x80000000u) ? ~u : (u | 0x80000000u);
}
__device__ __forceinline__ float decf(unsigned u) {
  unsigned b = (u & 0x80000000u) ? (u & 0x7FFFFFFFu) : ~u;
  return __uint_as_float(b);
}

// ---------------------------------------------------------------------------
// Pack a (K x 64) row-major f32 weight matrix into WMMA-B per-lane bf16 tiles.
// ---------------------------------------------------------------------------
__global__ void k_pack_b(const float* __restrict__ src, bf16_t* __restrict__ dst,
                         int K) {
  int tid = blockIdx.x * blockDim.x + threadIdx.x;
  int total = (K / 32) * 4 * 32;
  if (tid >= total) return;
  int lane = tid & 31;
  int nt = (tid >> 5) & 3;
  int kt = tid >> 7;
  bf16_t* out = dst + ((kt * 4 + nt) * 32 + lane) * 16;
  int n = nt * 16 + (lane & 15);
  int kb = kt * 32 + 8 * (lane >> 4);
#pragma unroll
  for (int e = 0; e < 16; ++e) out[e] = (bf16_t)src[(kb + e + (e >= 8 ? 8 : 0)) * HID + n];
}

__global__ void k_init_h(const float* __restrict__ in_w, const float* __restrict__ in_b,
                         float* __restrict__ hf, bf16_t* __restrict__ hb,
                         long long total) {
  long long tid = (long long)blockIdx.x * blockDim.x + threadIdx.x;
  if (tid >= total) return;
  int j = (int)(tid & (HID - 1));
  float v = in_w[j] + in_b[j];
  hf[tid] = v;
  hb[tid] = (bf16_t)v;
}

// ---------------------------------------------------------------------------
// Edge kernel: block stages packed w1/w2 into LDS (async), then each wave
// processes TILES x 16 edges, B fragments served from LDS.
// ---------------------------------------------------------------------------
__global__ void __launch_bounds__(256)
k_edge_layer(const long long* __restrict__ esrc, const long long* __restrict__ edst,
             const float* __restrict__ ef, const bf16_t* __restrict__ h_bf,
             const bf16_t* __restrict__ w1p, const float* __restrict__ w1last,
             const float* __restrict__ b1, const bf16_t* __restrict__ w2p,
             const float* __restrict__ b2, float* __restrict__ agg, long long E) {
  __shared__ __align__(16) bf16_t sw1[128 * HID];           // 16 KB
  __shared__ __align__(16) bf16_t sw2[64 * HID];            // 8 KB
  __shared__ __align__(16) bf16_t lds_act[8][16 * HID];     // 16 KB
  const int lane = threadIdx.x & 31;
  const int wib = threadIdx.x >> 5;
  const int row = lane & 15;
  const int half = lane >> 4;

  // async-stage weights into LDS
#pragma unroll
  for (int i = threadIdx.x; i < 1024; i += 256)
    async_copy16((char*)sw1 + i * 16, (const char*)w1p + i * 16);
#pragma unroll
  for (int i = threadIdx.x; i < 512; i += 256)
    async_copy16((char*)sw2 + i * 16, (const char*)w2p + i * 16);
  wait_async0();
  __syncthreads();

  // hoisted per-lane constants (per n-tile column)
  float b1c[4], wlc[4], b2c[4];
#pragma unroll
  for (int nt = 0; nt < 4; ++nt) {
    int c = nt * 16 + row;
    b1c[nt] = b1[c];
    wlc[nt] = w1last[c];
    b2c[nt] = b2[c];
  }

  const long long block_base = (long long)blockIdx.x * (8LL * TILES * 16);
  bf16_t* sact = lds_act[wib];

  for (int t = 0; t < TILES; ++t) {
    const long long ebase = block_base + (long long)(wib * TILES + t) * 16;
    // prefetch next tile's index/feature stream
    if (t + 1 < TILES) {
      __builtin_prefetch(esrc + ebase + 16, 0, 0);
      __builtin_prefetch(edst + ebase + 16, 0, 0);
      __builtin_prefetch(ef + ebase + 16, 0, 0);
    }
    long long erow = ebase + row;
    if (erow >= E) erow = E - 1;
    const int src = (int)esrc[erow];
    const int dst = (int)edst[erow];
    const bf16_t* hsrc = h_bf + (long long)src * HID;
    const bf16_t* hdst = h_bf + (long long)dst * HID;

    f32x8 acc[4];
#pragma unroll
    for (int q = 0; q < 4; ++q)
#pragma unroll
      for (int i = 0; i < 8; ++i) acc[q][i] = 0.0f;

    // GEMM1: K=128 ([h_src | h_dst]), B from LDS
#pragma unroll
    for (int kt = 0; kt < 4; ++kt) {
      const bf16_t* rowp = (kt < 2) ? hsrc : hdst;
      bf16x16 a = load_a_row(rowp, (kt & 1) * 32, half);
#pragma unroll
      for (int nt = 0; nt < 4; ++nt) {
        bf16x16 b = load_b_packed(sw1 + (kt * 4 + nt) * 512, lane);
        acc[nt] = wmma_bf16(a, b, acc[nt]);
      }
    }

    // edge-feature rank-1 term + bias + SiLU -> LDS (bf16, row-major)
    float efv[8];
#pragma unroll
    for (int r = 0; r < 8; ++r) {
      long long e2 = ebase + 8 * half + r;
      efv[r] = ef[e2 < E ? e2 : E - 1];
    }
#pragma unroll
    for (int nt = 0; nt < 4; ++nt) {
      int c = nt * 16 + row;
#pragma unroll
      for (int r = 0; r < 8; ++r) {
        float x = acc[nt][r] + b1c[nt] + efv[r] * wlc[nt];
        sact[(r + 8 * half) * HID + c] = (bf16_t)silu_f(x);
      }
    }
    __syncthreads();

    // GEMM2: m = act(16x64) @ w2(64x64)
    f32x8 acc2[4];
#pragma unroll
    for (int q = 0; q < 4; ++q)
#pragma unroll
      for (int i = 0; i < 8; ++i) acc2[q][i] = 0.0f;
#pragma unroll
    for (int kt = 0; kt < 2; ++kt) {
      bf16x16 a = load_a_row(sact + row * HID, kt * 32, half);
#pragma unroll
      for (int nt = 0; nt < 4; ++nt) {
        bf16x16 b = load_b_packed(sw2 + (kt * 4 + nt) * 512, lane);
        acc2[nt] = wmma_bf16(a, b, acc2[nt]);
      }
    }

    // scatter-add into agg
    int d8[8];
    bool v8[8];
#pragma unroll
    for (int r = 0; r < 8; ++r) {
      long long e2 = ebase + 8 * half + r;
      v8[r] = e2 < E;
      d8[r] = (int)edst[v8[r] ? e2 : E - 1];
    }
#pragma unroll
    for (int nt = 0; nt < 4; ++nt) {
      int c = nt * 16 + row;
#pragma unroll
      for (int r = 0; r < 8; ++r) {
        if (v8[r]) atomicAdd(agg + (long long)d8[r] * HID + c, acc2[nt][r] + b2c[nt]);
      }
    }
    __syncthreads();
  }
}

// ---------------------------------------------------------------------------
// Node kernel: u = silu([h|agg] @ u1 + c1) @ u2 + c2 ; h += u
// ---------------------------------------------------------------------------
__global__ void __launch_bounds__(256)
k_node_layer(bf16_t* h_bf, float* h_f, const float* __restrict__ agg,
             const bf16_t* __restrict__ u1p, const float* __restrict__ c1,
             const bf16_t* __restrict__ u2p, const float* __restrict__ c2, int N) {
  __shared__ __align__(16) bf16_t sw1[128 * HID];
  __shared__ __align__(16) bf16_t sw2[64 * HID];
  __shared__ __align__(16) bf16_t lds_act[8][16 * HID];
  const int lane = threadIdx.x & 31;
  const int wib = threadIdx.x >> 5;
  const int row = lane & 15;
  const int half = lane >> 4;

#pragma unroll
  for (int i = threadIdx.x; i < 1024; i += 256)
    async_copy16((char*)sw1 + i * 16, (const char*)u1p + i * 16);
#pragma unroll
  for (int i = threadIdx.x; i < 512; i += 256)
    async_copy16((char*)sw2 + i * 16, (const char*)u2p + i * 16);
  wait_async0();
  __syncthreads();

  float c1c[4], c2c[4];
#pragma unroll
  for (int nt = 0; nt < 4; ++nt) {
    int c = nt * 16 + row;
    c1c[nt] = c1[c];
    c2c[nt] = c2[c];
  }

  const int block_base = blockIdx.x * (8 * TILES * 16);
  bf16_t* sact = lds_act[wib];

  for (int t = 0; t < TILES; ++t) {
    const int nodebase = block_base + (wib * TILES + t) * 16;
    int node = nodebase + row;
    int nclamp = node < N ? node : N - 1;
    const bf16_t* hrow = h_bf + (long long)nclamp * HID;

    f32x8 acc[4];
#pragma unroll
    for (int q = 0; q < 4; ++q)
#pragma unroll
      for (int i = 0; i < 8; ++i) acc[q][i] = 0.0f;

#pragma unroll
    for (int kt = 0; kt < 4; ++kt) {
      bf16x16 a;
      if (kt < 2) {
        a = load_a_row(hrow, kt * 32, half);
      } else {
        const float* arow = agg + (long long)nclamp * HID + (kt - 2) * 32;
#pragma unroll
        for (int i = 0; i < 8; ++i) a[i] = (bf16_t)arow[8 * half + i];
#pragma unroll
        for (int i = 0; i < 8; ++i) a[i + 8] = (bf16_t)arow[16 + 8 * half + i];
      }
#pragma unroll
      for (int nt = 0; nt < 4; ++nt) {
        bf16x16 b = load_b_packed(sw1 + (kt * 4 + nt) * 512, lane);
        acc[nt] = wmma_bf16(a, b, acc[nt]);
      }
    }

#pragma unroll
    for (int nt = 0; nt < 4; ++nt) {
      int c = nt * 16 + row;
#pragma unroll
      for (int r = 0; r < 8; ++r)
        sact[(r + 8 * half) * HID + c] = (bf16_t)silu_f(acc[nt][r] + c1c[nt]);
    }
    __syncthreads();

    f32x8 acc2[4];
#pragma unroll
    for (int q = 0; q < 4; ++q)
#pragma unroll
      for (int i = 0; i < 8; ++i) acc2[q][i] = 0.0f;
#pragma unroll
    for (int kt = 0; kt < 2; ++kt) {
      bf16x16 a = load_a_row(sact + row * HID, kt * 32, half);
#pragma unroll
      for (int nt = 0; nt < 4; ++nt) {
        bf16x16 b = load_b_packed(sw2 + (kt * 4 + nt) * 512, lane);
        acc2[nt] = wmma_bf16(a, b, acc2[nt]);
      }
    }

#pragma unroll
    for (int nt = 0; nt < 4; ++nt) {
      int c = nt * 16 + row;
#pragma unroll
      for (int r = 0; r < 8; ++r) {
        int nr = nodebase + 8 * half + r;
        if (nr < N) {
          long long idx = (long long)nr * HID + c;
          float v = h_f[idx] + acc2[nt][r] + c2c[nt];
          h_f[idx] = v;
          h_bf[idx] = (bf16_t)v;
        }
      }
    }
    __syncthreads();
  }
}

// ---------------------------------------------------------------------------
// Readout: logits = silu(h @ ro_w1 + ro_b1) @ ro_w2 + ro_b2 ; track global max
// ---------------------------------------------------------------------------
__global__ void __launch_bounds__(256)
k_readout(const bf16_t* __restrict__ h_bf, const bf16_t* __restrict__ ro1p,
          const float* __restrict__ rb1, const float* __restrict__ ro_w2,
          const float* __restrict__ rb2, float* __restrict__ logits,
          unsigned* __restrict__ maxenc, int N) {
  __shared__ __align__(16) bf16_t sw1[64 * HID];            // 8 KB
  __shared__ __align__(16) float lds_act[8][16 * HID];      // 32 KB
  const int lane = threadIdx.x & 31;
  const int wib = threadIdx.x >> 5;
  const int row = lane & 15;
  const int half = lane >> 4;

#pragma unroll
  for (int i = threadIdx.x; i < 512; i += 256)
    async_copy16((char*)sw1 + i * 16, (const char*)ro1p + i * 16);
  wait_async0();
  __syncthreads();

  float rb1c[4];
#pragma unroll
  for (int nt = 0; nt < 4; ++nt) rb1c[nt] = rb1[nt * 16 + row];

  const int block_base = blockIdx.x * (8 * TILES * 16);
  float* sact = lds_act[wib];

  for (int t = 0; t < TILES; ++t) {
    const int nodebase = block_base + (wib * TILES + t) * 16;
    int node = nodebase + row;
    int nclamp = node < N ? node : N - 1;
    const bf16_t* hrow = h_bf + (long long)nclamp * HID;

    f32x8 acc[4];
#pragma unroll
    for (int q = 0; q < 4; ++q)
#pragma unroll
      for (int i = 0; i < 8; ++i) acc[q][i] = 0.0f;

#pragma unroll
    for (int kt = 0; kt < 2; ++kt) {
      bf16x16 a = load_a_row(hrow, kt * 32, half);
#pragma unroll
      for (int nt = 0; nt < 4; ++nt) {
        bf16x16 b = load_b_packed(sw1 + (kt * 4 + nt) * 512, lane);
        acc[nt] = wmma_bf16(a, b, acc[nt]);
      }
    }

#pragma unroll
    for (int nt = 0; nt < 4; ++nt) {
      int c = nt * 16 + row;
#pragma unroll
      for (int r = 0; r < 8; ++r)
        sact[(r + 8 * half) * HID + c] = silu_f(acc[nt][r] + rb1c[nt]);
    }
    __syncthreads();

    if (lane < 16) {
      int n2 = nodebase + lane;
      if (n2 < N) {
        const float* ar = sact + lane * HID;
        float dot = rb2[0];
#pragma unroll
        for (int j = 0; j < HID; ++j) dot += ar[j] * ro_w2[j];
        logits[n2] = dot;
        atomicMax(maxenc, encf(dot));
      }
    }
    __syncthreads();
  }
}

__global__ void k_expsum(float* __restrict__ logits, const unsigned* __restrict__ maxenc,
                         float* __restrict__ sum, int N) {
  __shared__ float red[256];
  int tid = blockIdx.x * blockDim.x + threadIdx.x;
  float e = 0.0f;
  if (tid < N) {
    float m = decf(*maxenc);
    e = __expf(logits[tid] - m);
    logits[tid] = e;
  }
  red[threadIdx.x] = e;
  __syncthreads();
#pragma unroll
  for (int s = 128; s > 0; s >>= 1) {
    if ((int)threadIdx.x < s) red[threadIdx.x] += red[threadIdx.x + s];
    __syncthreads();
  }
  if (threadIdx.x == 0) atomicAdd(sum, red[0]);
}

__global__ void k_norm(const float* __restrict__ logits, const float* __restrict__ sum,
                       float* __restrict__ out, int N) {
  int tid = blockIdx.x * blockDim.x + threadIdx.x;
  if (tid < N) out[tid] = logits[tid] * (1.0f / *sum);
}

// ---------------------------------------------------------------------------
extern "C" void kernel_launch(void* const* d_in, const int* in_sizes, int n_in,
                              void* d_out, int out_size, void* d_ws, size_t ws_size,
                              hipStream_t stream) {
  (void)n_in; (void)out_size; (void)ws_size;
  const long long* ei = (const long long*)d_in[0];
  const float* ef = (const float*)d_in[1];
  const float* in_w = (const float*)d_in[3];
  const float* in_b = (const float*)d_in[4];
  const float* msg_w1 = (const float*)d_in[5];
  const float* msg_b1 = (const float*)d_in[6];
  const float* msg_w2 = (const float*)d_in[7];
  const float* msg_b2 = (const float*)d_in[8];
  const float* upd_w1 = (const float*)d_in[9];
  const float* upd_b1 = (const float*)d_in[10];
  const float* upd_w2 = (const float*)d_in[11];
  const float* upd_b2 = (const float*)d_in[12];
  const float* ro_w1 = (const float*)d_in[13];
  const float* ro_b1 = (const float*)d_in[14];
  const float* ro_w2 = (const float*)d_in[15];
  const float* ro_b2 = (const float*)d_in[16];

  const long long E = (long long)in_sizes[0] / 2;
  const int N = N_NODES_C;

  char* ws = (char*)d_ws;
  size_t off = 0;
  auto carve = [&](size_t bytes) -> void* {
    void* p = ws + off;
    off += (bytes + 255) & ~(size_t)255;
    return p;
  };
  float* hf = (float*)carve((size_t)N * HID * sizeof(float));
  bf16_t* hb = (bf16_t*)carve((size_t)N * HID * sizeof(bf16_t));
  float* agg = (float*)carve((size_t)N * HID * sizeof(float));
  float* logits = (float*)carve((size_t)N * sizeof(float));
  unsigned* maxenc = (unsigned*)carve(256);
  float* sum = (float*)carve(256);
  bf16_t* pw_m1 = (bf16_t*)carve((size_t)NLAYERS * 128 * HID * sizeof(bf16_t));
  bf16_t* pw_m2 = (bf16_t*)carve((size_t)NLAYERS * 64 * HID * sizeof(bf16_t));
  bf16_t* pw_u1 = (bf16_t*)carve((size_t)NLAYERS * 128 * HID * sizeof(bf16_t));
  bf16_t* pw_u2 = (bf16_t*)carve((size_t)NLAYERS * 64 * HID * sizeof(bf16_t));
  bf16_t* pw_r1 = (bf16_t*)carve((size_t)64 * HID * sizeof(bf16_t));

  {
    long long total = (long long)N * HID;
    int blocks = (int)((total + 255) / 256);
    k_init_h<<<blocks, 256, 0, stream>>>(in_w, in_b, hf, hb, total);
  }
  for (int l = 0; l < NLAYERS; ++l) {
    k_pack_b<<<2, 256, 0, stream>>>(msg_w1 + (size_t)l * 129 * HID,
                                    pw_m1 + (size_t)l * 128 * HID, 128);
    k_pack_b<<<1, 256, 0, stream>>>(msg_w2 + (size_t)l * 64 * HID,
                                    pw_m2 + (size_t)l * 64 * HID, 64);
    k_pack_b<<<2, 256, 0, stream>>>(upd_w1 + (size_t)l * 128 * HID,
                                    pw_u1 + (size_t)l * 128 * HID, 128);
    k_pack_b<<<1, 256, 0, stream>>>(upd_w2 + (size_t)l * 64 * HID,
                                    pw_u2 + (size_t)l * 64 * HID, 64);
  }
  k_pack_b<<<1, 256, 0, stream>>>(ro_w1, pw_r1, 64);

  const int rows_per_block = 8 * TILES * 16;  // 512
  const int edge_blocks = (int)((E + rows_per_block - 1) / rows_per_block);
  const int node_blocks = (N + rows_per_block - 1) / rows_per_block;

  for (int l = 0; l < NLAYERS; ++l) {
    hipMemsetAsync(agg, 0, (size_t)N * HID * sizeof(float), stream);
    k_edge_layer<<<edge_blocks, 256, 0, stream>>>(
        ei, ei + E, ef, hb, pw_m1 + (size_t)l * 128 * HID,
        msg_w1 + (size_t)l * 129 * HID + 128 * HID, msg_b1 + (size_t)l * HID,
        pw_m2 + (size_t)l * 64 * HID, msg_b2 + (size_t)l * HID, agg, E);
    k_node_layer<<<node_blocks, 256, 0, stream>>>(
        hb, hf, agg, pw_u1 + (size_t)l * 128 * HID, upd_b1 + (size_t)l * HID,
        pw_u2 + (size_t)l * 64 * HID, upd_b2 + (size_t)l * HID, N);
  }

  hipMemsetAsync(maxenc, 0, sizeof(unsigned), stream);
  hipMemsetAsync(sum, 0, sizeof(float), stream);
  k_readout<<<node_blocks, 256, 0, stream>>>(hb, pw_r1, ro_b1, ro_w2, ro_b2, logits,
                                             maxenc, N);
  const int nb = (N + 255) / 256;
  k_expsum<<<nb, 256, 0, stream>>>(logits, maxenc, sum, N);
  k_norm<<<nb, 256, 0, stream>>>(logits, sum, (float*)d_out, N);
}
